// GPT_24799141167440
// MI455X (gfx1250) — matrix operand order repaired and checked by reading, not compiled, not run on hardware
//
#include <hip/hip_runtime.h>
#include <math.h>

// ---------------- CDNA5 WMMA types ----------------
typedef __attribute__((ext_vector_type(16))) __bf16 v16bf;
typedef __attribute__((ext_vector_type(8)))  __bf16 v8bf;
typedef __attribute__((ext_vector_type(4)))  __bf16 v4bf;
typedef __attribute__((ext_vector_type(8)))  float  v8f;
typedef __attribute__((ext_vector_type(4)))  int    v4i;

// Async global->LDS path (CDNA5), compile-time guarded with clean fallback.
#if defined(__gfx1250__) && __has_builtin(__builtin_amdgcn_global_load_async_to_lds_b128) && __has_builtin(__builtin_amdgcn_s_wait_asynccnt)
#define USE_ASYNC_LDS 1
typedef __attribute__((address_space(1))) v4i* as1_v4i;
typedef __attribute__((address_space(3))) v4i* as3_v4i;
#else
#define USE_ASYNC_LDS 0
#endif

// Native conversion: gfx1250 emits v_cvt_pk_bf16_f32 (confirmed round 3).
__device__ inline __bf16 f2bf(float f) { return (__bf16)f; }

__device__ inline v4bf f4tobf(float4 f) {
  v4bf r; r[0] = f2bf(f.x); r[1] = f2bf(f.y); r[2] = f2bf(f.z); r[3] = f2bf(f.w);
  return r;
}
__device__ inline v8bf cat4(v4bf lo, v4bf hi) {
  return __builtin_shufflevector(lo, hi, 0, 1, 2, 3, 4, 5, 6, 7);
}
__device__ inline v16bf cat8(v8bf lo, v8bf hi) {
  return __builtin_shufflevector(lo, hi, 0, 1, 2, 3, 4, 5, 6, 7,
                                 8, 9, 10, 11, 12, 13, 14, 15);
}

// K-column permutation (swaps k bits 3 and 4) used for the A and P tiles in LDS.
// A-frag: lanes 0-15 read cols {0..7,16..23} = K 0..15, lanes 16-31 cols {8..15,24..31} = K 16..31.
__device__ inline int fcol(int k) {
  return (k & 7) | ((k & 8) << 1) | ((k & 16) >> 1);
}

__device__ inline float gelu_exact(float x) {
  return 0.5f * x * (1.0f + erff(x * 0.70710678118654752440f));
}

// ---------------- Embedding (fp32 residual stream) ----------------
__global__ __launch_bounds__(256) void embed_kernel(
    const int* __restrict__ tokens, const float* __restrict__ wte,
    const float* __restrict__ wpe, float* __restrict__ x, int S, int D) {
  int row = blockIdx.x;
  int s = row % S;
  int tok = tokens[row];
  for (int i = threadIdx.x; i < D; i += 256)
    x[(size_t)row * D + i] = wte[(size_t)tok * D + i] + wpe[(size_t)s * D + i];
}

// ---------------- LayerNorm: fp32 in, bf16 out ----------------
__global__ __launch_bounds__(256) void ln_kernel(
    const float* __restrict__ x, const float* __restrict__ g,
    const float* __restrict__ b, __bf16* __restrict__ out, int D) {
  int row = blockIdx.x;
  const float* xr = x + (size_t)row * D;
  __shared__ float red[256];
  int tid = threadIdx.x;

  float s = 0.f;
  for (int i = tid; i < D; i += 256) s += xr[i];
  red[tid] = s; __syncthreads();
  for (int o = 128; o > 0; o >>= 1) {
    if (tid < o) red[tid] += red[tid + o];
    __syncthreads();
  }
  float mu = red[0] / (float)D;
  __syncthreads();

  float v = 0.f;
  for (int i = tid; i < D; i += 256) { float d = xr[i] - mu; v += d * d; }
  red[tid] = v; __syncthreads();
  for (int o = 128; o > 0; o >>= 1) {
    if (tid < o) red[tid] += red[tid + o];
    __syncthreads();
  }
  float rstd = rsqrtf(red[0] / (float)D + 1e-5f);

  for (int i = tid; i < D; i += 256)
    out[(size_t)row * D + i] = f2bf((xr[i] - mu) * rstd * g[i] + b[i]);
}

// ---------------- bf16 WMMA GEMM, 64x128x32 tiles, double-buffered ----------------
// C[M,N] = act(A[M,K] @ B + bias) + resid.  A bf16 [M,K].
// B fp32 row-major [K,N] or, for BTRANS=1, [N,K].  C fp32 or bf16 (obf16).
// BTRANS is a template parameter: one clean staging path per instantiation.
#define BM 64
#define BN 128
#define BKK 32
#define LDP 40   // padded LDS row (80 bytes: 16B-aligned, bank-spread)

template <int BTRANS>
__global__ __launch_bounds__(256, 1) void gemm_kernel(
    const __bf16* __restrict__ A, const float* __restrict__ Bm,
    const float* __restrict__ bias, const float* resid,
    void* Cout, int M, int N, int K, int act, int obf16) {
  __shared__ __attribute__((aligned(16))) __bf16 Alds[2][BM][LDP];  // K-permuted
  __shared__ __attribute__((aligned(16))) __bf16 Blds[2][BN][LDP];  // frag-major [n][k]

  const int tid  = threadIdx.x;
  const int lane = tid & 31;
  const int wave = tid >> 5;
  const int row0 = (wave & 3) * 16;   // wave's 16-row strip
  const int col0 = (wave >> 2) * 64;  // wave's 64-col strip (4 frags)

  const int tileM = blockIdx.y * BM;
  const int tileN = blockIdx.x * BN;

  const int m_lane = lane & 15;
  const int khalf  = lane >> 4;

  // staging coordinates
  const int arow  = tid >> 2;          // A row 0..63 (bf16: 1 x 16B per thread)
  const int ac8   = (tid & 3) * 8;     // A k-group {0,8,16,24}
  const int afc8  = fcol(ac8);
  const int btrow = tid >> 3;          // B^T n (and +32*p)
  const int ac    = (tid & 7) * 4;     // B^T k sub-offset 0..28 (float4)
  const int kb    = (tid >> 5) * 4;    // B normal: 4 k-rows
  const int nb    = (tid & 31) * 4;    // B normal: 4 n-cols

  float4 bv[4];
#if !USE_ASYNC_LDS
  v8bf areg;
#endif

#if USE_ASYNC_LDS
  auto stageA = [&](int k0, int nbuf) {   // async byte-mover: no VGPR round trip
    const __bf16* src = &A[(size_t)(tileM + arow) * K + k0 + ac8];
    __builtin_amdgcn_global_load_async_to_lds_b128(
        (as1_v4i)src, (as3_v4i)&Alds[nbuf][arow][afc8], 0, 0);
  };
#else
  auto loadA = [&](int k0) {
    areg = *(const v8bf*)&A[(size_t)(tileM + arow) * K + k0 + ac8];
  };
  auto storeA = [&](int nbuf) {
    *(v8bf*)&Alds[nbuf][arow][afc8] = areg;
  };
#endif
  auto loadB = [&](int k0) {
    if (!BTRANS) {                    // [K,N]; N is a multiple of BN here
      #pragma unroll
      for (int i = 0; i < 4; ++i)
        bv[i] = *(const float4*)&Bm[(size_t)(k0 + kb + i) * N + tileN + nb];
    } else {                          // [N,K]; contiguous along k, guard N tail
      #pragma unroll
      for (int p = 0; p < 4; ++p) {
        int nn = tileN + btrow + p * 32;
        bv[p] = make_float4(0.f, 0.f, 0.f, 0.f);
        if (nn < N) bv[p] = *(const float4*)&Bm[(size_t)nn * K + k0 + ac];
      }
    }
  };
  auto storeB = [&](int nbuf) {
    if (!BTRANS) {                    // 4x4 register transpose -> ds_store_b64
      #pragma unroll
      for (int j = 0; j < 4; ++j) {
        v4bf t;
        #pragma unroll
        for (int i = 0; i < 4; ++i) t[i] = f2bf(((const float*)&bv[i])[j]);
        *(v4bf*)&Blds[nbuf][nb + j][kb] = t;
      }
    } else {
      #pragma unroll
      for (int p = 0; p < 4; ++p)
        *(v4bf*)&Blds[nbuf][btrow + p * 32][ac] = f4tobf(bv[p]);
    }
  };

  v8f acc[4];
  #pragma unroll
  for (int f = 0; f < 4; ++f)
    #pragma unroll
    for (int r = 0; r < 8; ++r) acc[f][r] = 0.f;

  // All 4 B frags preloaded first -> partial s_wait_dscnt, WMMA/load overlap.
  auto compute = [&](int cbuf) {
    const __bf16* ar = &Alds[cbuf][row0 + m_lane][0];
    v16bf a = cat8(*(const v8bf*)(ar + khalf * 8),
                   *(const v8bf*)(ar + khalf * 8 + 16));
    v16bf bfr[4];
    #pragma unroll
    for (int f = 0; f < 4; ++f) {
      const __bf16* br = &Blds[cbuf][col0 + f * 16 + m_lane][0];
      bfr[f] = cat8(*(const v8bf*)(br + khalf * 16),
                    *(const v8bf*)(br + khalf * 16 + 8));
    }
    #pragma unroll
    for (int f = 0; f < 4; ++f)
      acc[f] = __builtin_amdgcn_wmma_f32_16x16x32_bf16(false, a, false, bfr[f],
                                                       (short)0, acc[f], false, false);
  };

  // prologue: stage tile 0
#if USE_ASYNC_LDS
  stageA(0, 0);
#else
  loadA(0);
#endif
  loadB(0);
#if !USE_ASYNC_LDS
  storeA(0);
#endif
  storeB(0);

  // steady state: unconditional stage of next tile + compute of current
  int buf = 0;
  int k0 = 0;
  for (; k0 + BKK < K; k0 += BKK) {
#if USE_ASYNC_LDS
    __builtin_amdgcn_s_wait_asynccnt(0);   // publish this wave's async A tile
#endif
    __syncthreads();
#if USE_ASYNC_LDS
    stageA(k0 + BKK, buf ^ 1);
#else
    loadA(k0 + BKK);
#endif
    loadB(k0 + BKK);                       // overlap with WMMA below

    compute(buf);

#if !USE_ASYNC_LDS
    storeA(buf ^ 1);
#endif
    storeB(buf ^ 1);
    buf ^= 1;
  }
  // peeled last iteration: compute only
#if USE_ASYNC_LDS
  __builtin_amdgcn_s_wait_asynccnt(0);
#endif
  __syncthreads();
  compute(buf);

  // ---- epilogue ----
  float*  Cf = (float*)Cout;
  __bf16* Cb = (__bf16*)Cout;
  #pragma unroll
  for (int r = 0; r < 8; ++r) {
    int gm = tileM + row0 + r + khalf * 8;
    #pragma unroll
    for (int f = 0; f < 4; ++f) {
      int gn = tileN + col0 + f * 16 + m_lane;
      if (gn < N) {
        float v = acc[f][r];
        if (bias)  v += bias[gn];
        if (act)   v = gelu_exact(v);
        if (resid) v += resid[(size_t)gm * N + gn];
        if (obf16) Cb[(size_t)gm * N + gn] = f2bf(v);
        else       Cf[(size_t)gm * N + gn] = v;
      }
    }
  }
}

// ---------------- Fused causal flash attention (bf16 QKV in, bf16 out) ----------------
// One wave32 per (batch, head, 16-query tile). scale = 1/sqrt(D) (ref quirk).
__global__ __launch_bounds__(32) void attn_kernel(
    const __bf16* __restrict__ qkv, __bf16* __restrict__ out,
    int S, int DHn, int Dn, float scale) {
  const int qt   = blockIdx.x;
  const int head = blockIdx.y;
  const int bb   = blockIdx.z;
  const int q0   = qt * 16;
  const int lane = threadIdx.x;
  const int m_lane = lane & 15;
  const int khalf  = lane >> 4;
  const int TD = 3 * Dn;

  const __bf16* qp = qkv + (size_t)bb * S * TD + head * DHn;
  const __bf16* kp = qp + Dn;
  const __bf16* vp = qp + 2 * Dn;

  __shared__ __attribute__((aligned(16))) __bf16 Plds[16][LDP];  // K-permuted P
  __shared__ __attribute__((aligned(16))) __bf16 Vt[64][LDP];    // V^T [d][key]

  // Q fragments: two contiguous 16B runs per 32-d chunk
  v16bf qf[2];
  const size_t qbase = (size_t)(q0 + m_lane) * TD;
  #pragma unroll
  for (int c2 = 0; c2 < 2; ++c2) {
    const __bf16* qb = qp + qbase + c2 * 32 + khalf * 16;
    qf[c2] = cat8(*(const v8bf*)qb, *(const v8bf*)(qb + 8));
  }

  v8f Ofr[4];
  float mrow[8], lrow[8];
  #pragma unroll
  for (int r = 0; r < 8; ++r) {
    mrow[r] = -1e30f; lrow[r] = 0.f;
    Ofr[0][r] = 0.f; Ofr[1][r] = 0.f; Ofr[2][r] = 0.f; Ofr[3][r] = 0.f;
  }

  const int pg  = (m_lane & 8) << 1;           // P-column permutation group
  const int pc0 = (m_lane & 7) | pg;           // fcol(m_lane)
  const int pc1 = pc0 | 8;                     // fcol(16 + m_lane)

  const int vkb = (lane & 7) * 4;              // V staging: 4 keys
  const int vdg = (lane >> 3) * 16;            // V staging: 16 d values

  const int nchunks = (qt + 2) / 2;            // ceil((qt+1)/2), 32 keys/chunk
  for (int c = 0; c < nchunks; ++c) {
    const int k0 = c * 32;

    // ---- stage V chunk transposed: bf16 4x4 register transpose ----
    #pragma unroll
    for (int d4 = 0; d4 < 4; ++d4) {
      v4bf vv[4];
      #pragma unroll
      for (int i = 0; i < 4; ++i)
        vv[i] = *(const v4bf*)(vp + (size_t)(k0 + vkb + i) * TD + vdg + d4 * 4);
      #pragma unroll
      for (int j = 0; j < 4; ++j) {
        v4bf t;
        #pragma unroll
        for (int i = 0; i < 4; ++i) t[i] = vv[i][j];
        *(v4bf*)&Vt[vdg + d4 * 4 + j][vkb] = t;
      }
    }

    // ---- scores: two 16-key subtiles ----
    v8f sc[2];
    #pragma unroll
    for (int t = 0; t < 2; ++t) {
      const int kk0 = k0 + t * 16;
      #pragma unroll
      for (int r = 0; r < 8; ++r) sc[t][r] = 0.f;
      if (kk0 <= q0 + 15) {                    // block-uniform
        #pragma unroll
        for (int dch = 0; dch < 2; ++dch) {
          const __bf16* kbp = kp + (size_t)(kk0 + m_lane) * TD + dch * 32 + khalf * 16;
          v16bf bf = cat8(*(const v8bf*)kbp, *(const v8bf*)(kbp + 8));
          sc[t] = __builtin_amdgcn_wmma_f32_16x16x32_bf16(
              false, qf[dch], false, bf, (short)0, sc[t], false, false);
        }
        #pragma unroll
        for (int r = 0; r < 8; ++r) {
          int mq  = r + khalf * 8;
          int key = kk0 + m_lane;
          float v = sc[t][r] * scale;
          if (key > q0 + mq) v = -1e30f;       // causal mask
          sc[t][r] = v;
        }
      } else {
        #pragma unroll
        for (int r = 0; r < 8; ++r) sc[t][r] = -1e30f;
      }
    }

    // ---- online softmax (16-lane xor reductions per row) ----
    #pragma unroll
    for (int r = 0; r < 8; ++r) {
      float tmx = fmaxf(sc[0][r], sc[1][r]);
      tmx = fmaxf(tmx, __shfl_xor(tmx, 1, 32));
      tmx = fmaxf(tmx, __shfl_xor(tmx, 2, 32));
      tmx = fmaxf(tmx, __shfl_xor(tmx, 4, 32));
      tmx = fmaxf(tmx, __shfl_xor(tmx, 8, 32));
      float newm = fmaxf(mrow[r], tmx);
      float alpha = expf(mrow[r] - newm);
      mrow[r] = newm;
      float p0 = expf(sc[0][r] - newm);
      float p1 = expf(sc[1][r] - newm);
      sc[0][r] = p0; sc[1][r] = p1;
      float rs = p0 + p1;
      rs += __shfl_xor(rs, 1, 32);
      rs += __shfl_xor(rs, 2, 32);
      rs += __shfl_xor(rs, 4, 32);
      rs += __shfl_xor(rs, 8, 32);
      lrow[r] = lrow[r] * alpha + rs;
      Ofr[0][r] *= alpha; Ofr[1][r] *= alpha;
      Ofr[2][r] *= alpha; Ofr[3][r] *= alpha;
    }

    // ---- P -> LDS (K-permuted columns for contiguous A-frag reload) ----
    #pragma unroll
    for (int r = 0; r < 8; ++r) {
      int mq = r + khalf * 8;
      Plds[mq][pc0] = f2bf(sc[0][r]);
      Plds[mq][pc1] = f2bf(sc[1][r]);
    }
    __syncthreads();

    const __bf16* pr = &Plds[m_lane][0];
    v16bf pa = cat8(*(const v8bf*)(pr + khalf * 8),
                    *(const v8bf*)(pr + khalf * 8 + 16));
    // ---- O += P @ V : four b128-pair fragments from V^T ----
    #pragma unroll
    for (int f = 0; f < 4; ++f) {
      const __bf16* vr = &Vt[f * 16 + m_lane][0];
      v16bf vb = cat8(*(const v8bf*)(vr + khalf * 16),
                      *(const v8bf*)(vr + khalf * 16 + 8));
      Ofr[f] = __builtin_amdgcn_wmma_f32_16x16x32_bf16(
          false, pa, false, vb, (short)0, Ofr[f], false, false);
    }
    __syncthreads();
  }

  // ---- epilogue: O /= l, bf16 out ----
  #pragma unroll
  for (int r = 0; r < 8; ++r) {
    int mq = r + khalf * 8;
    size_t rowbase = (size_t)bb * S * Dn + (size_t)(q0 + mq) * Dn + head * DHn;
    float inv = 1.0f / lrow[r];
    #pragma unroll
    for (int f = 0; f < 4; ++f)
      out[rowbase + f * 16 + m_lane] = f2bf(Ofr[f][r] * inv);
  }
}

// ---------------- Row log-softmax (in place) ----------------
__global__ __launch_bounds__(256) void lsm_kernel(float* __restrict__ logits, int N) {
  size_t row = blockIdx.x;
  float* r = logits + row * (size_t)N;
  __shared__ float red[256];
  int tid = threadIdx.x;

  float mx = -1e30f;
  #pragma unroll 4
  for (int i = tid; i < N; i += 256) mx = fmaxf(mx, r[i]);
  red[tid] = mx; __syncthreads();
  for (int o = 128; o > 0; o >>= 1) {
    if (tid < o) red[tid] = fmaxf(red[tid], red[tid + o]);
    __syncthreads();
  }
  float m = red[0]; __syncthreads();

  float s = 0.f;
  #pragma unroll 4
  for (int i = tid; i < N; i += 256) s += expf(r[i] - m);
  red[tid] = s; __syncthreads();
  for (int o = 128; o > 0; o >>= 1) {
    if (tid < o) red[tid] += red[tid + o];
    __syncthreads();
  }
  float lse = m + logf(red[0]);
  #pragma unroll 4
  for (int i = tid; i < N; i += 256) r[i] = r[i] - lse;
}

// ---------------- Per-row CE with second log-softmax (ref quirk) ----------------
__global__ __launch_bounds__(256) void rowloss_kernel(
    const float* __restrict__ lp, const int* __restrict__ target,
    float* __restrict__ rowloss, int N) {
  size_t row = blockIdx.x;
  const float* r = lp + row * (size_t)N;
  __shared__ float red[256];
  int tid = threadIdx.x;

  float mx = -1e30f;
  #pragma unroll 4
  for (int i = tid; i < N; i += 256) mx = fmaxf(mx, r[i]);
  red[tid] = mx; __syncthreads();
  for (int o = 128; o > 0; o >>= 1) {
    if (tid < o) red[tid] = fmaxf(red[tid], red[tid + o]);
    __syncthreads();
  }
  float m = red[0]; __syncthreads();

  float s = 0.f;
  #pragma unroll 4
  for (int i = tid; i < N; i += 256) s += expf(r[i] - m);
  red[tid] = s; __syncthreads();
  for (int o = 128; o > 0; o >>= 1) {
    if (tid < o) red[tid] += red[tid + o];
    __syncthreads();
  }
  float lse = m + logf(red[0]);
  if (tid == 0) rowloss[row] = -(r[target[row]] - lse);
}

__global__ __launch_bounds__(256) void finloss_kernel(
    const float* __restrict__ rowloss, float* __restrict__ out, int M) {
  __shared__ float red[256];
  int tid = threadIdx.x;
  float s = 0.f;
  for (int i = tid; i < M; i += 256) s += rowloss[i];
  red[tid] = s; __syncthreads();
  for (int o = 128; o > 0; o >>= 1) {
    if (tid < o) red[tid] += red[tid + o];
    __syncthreads();
  }
  if (tid == 0) out[0] = red[0] / (float)M;
}

// ---------------- Host orchestration ----------------
extern "C" void kernel_launch(void* const* d_in, const int* in_sizes, int n_in,
                              void* d_out, int out_size, void* d_ws, size_t ws_size,
                              hipStream_t stream) {
  (void)in_sizes; (void)n_in; (void)ws_size;

  const int Bb = 2, S = 1024, Dm = 768, Hh = 12, DHh = 64, Ll = 12, FFf = 3072;
  const int Vv = 50257;
  const int M = Bb * S;
  const float scale = 1.0f / sqrtf((float)Dm);   // ref quirk: 1/sqrt(embed_dim)

  const int*   tokens = (const int*)  d_in[0];
  const int*   target = (const int*)  d_in[1];
  const float* wte    = (const float*)d_in[2];
  const float* wpe    = (const float*)d_in[3];
  const float* ln1_g  = (const float*)d_in[4];
  const float* ln1_b  = (const float*)d_in[5];
  const float* qkv_w  = (const float*)d_in[6];
  const float* qkv_b  = (const float*)d_in[7];
  const float* proj_w = (const float*)d_in[8];
  const float* proj_b = (const float*)d_in[9];
  const float* ln2_g  = (const float*)d_in[10];
  const float* ln2_b  = (const float*)d_in[11];
  const float* fc1_w  = (const float*)d_in[12];
  const float* fc1_b  = (const float*)d_in[13];
  const float* fc2_w  = (const float*)d_in[14];
  const float* fc2_b  = (const float*)d_in[15];
  const float* lnf_g  = (const float*)d_in[16];
  const float* lnf_b  = (const float*)d_in[17];

  // workspace carve (bf16 activations, fp32 residual stream)
  char* wsb = (char*)d_ws;
  float*  x    = (float*)wsb;   wsb += (size_t)M * Dm * 4;
  __bf16* hbf  = (__bf16*)wsb;  wsb += (size_t)M * Dm * 2;
  __bf16* qkvb = (__bf16*)wsb;  wsb += (size_t)M * 3 * Dm * 2;
  __bf16* attb = (__bf16*)wsb;  wsb += (size_t)M * Dm * 2;
  __bf16* ff1b = (__bf16*)wsb;  wsb += (size_t)M * FFf * 2;
  float*  rls  = (float*)wsb;

  float* logits = (float*)d_out;                 // [M, V] -> log-probs

  embed_kernel<<<M, 256, 0, stream>>>(tokens, wte, wpe, x, S, Dm);

  for (int l = 0; l < Ll; ++l) {
    ln_kernel<<<M, 256, 0, stream>>>(x, ln1_g + (size_t)l * Dm,
                                     ln1_b + (size_t)l * Dm, hbf, Dm);
    gemm_kernel<0><<<dim3((3 * Dm) / BN, M / BM), 256, 0, stream>>>(
        hbf, qkv_w + (size_t)l * Dm * 3 * Dm, qkv_b + (size_t)l * 3 * Dm,
        nullptr, qkvb, M, 3 * Dm, Dm, 0, 1);             // bf16 out
    attn_kernel<<<dim3(S / 16, Hh, Bb), 32, 0, stream>>>(qkvb, attb, S, DHh, Dm, scale);
    gemm_kernel<0><<<dim3(Dm / BN, M / BM), 256, 0, stream>>>(
        attb, proj_w + (size_t)l * Dm * Dm, proj_b + (size_t)l * Dm,
        x, x, M, Dm, Dm, 0, 0);                          // x += att@Wp + bp (fp32)
    ln_kernel<<<M, 256, 0, stream>>>(x, ln2_g + (size_t)l * Dm,
                                     ln2_b + (size_t)l * Dm, hbf, Dm);
    gemm_kernel<0><<<dim3(FFf / BN, M / BM), 256, 0, stream>>>(
        hbf, fc1_w + (size_t)l * Dm * FFf, fc1_b + (size_t)l * FFf,
        nullptr, ff1b, M, FFf, Dm, 1, 1);                // GELU, bf16 out
    gemm_kernel<0><<<dim3(Dm / BN, M / BM), 256, 0, stream>>>(
        ff1b, fc2_w + (size_t)l * FFf * Dm, fc2_b + (size_t)l * Dm,
        x, x, M, Dm, FFf, 0, 0);                         // x += mlp (fp32)
  }

  ln_kernel<<<M, 256, 0, stream>>>(x, lnf_g, lnf_b, hbf, Dm);
  gemm_kernel<1><<<dim3((Vv + BN - 1) / BN, M / BM), 256, 0, stream>>>(
      hbf, wte, nullptr, nullptr, logits, M, Vv, Dm, 0, 0);  // tied head, B^T

  lsm_kernel<<<M, 256, 0, stream>>>(logits, Vv);
  rowloss_kernel<<<M, 256, 0, stream>>>(logits, target, rls, Vv);
  finloss_kernel<<<1, 256, 0, stream>>>(rls, logits + (size_t)(out_size - 1), M);
}
